// DeepFM_21732534518459
// MI455X (gfx1250) — matrix operand order repaired and checked by reading, not compiled, not run on hardware
//
#include <hip/hip_runtime.h>
#include <hip/hip_bf16.h>

typedef __attribute__((ext_vector_type(16))) __bf16 v16bf;
typedef __attribute__((ext_vector_type(8)))  float  v8f;

#define BSZ   16384
#define FN    26
#define VN    100000
#define EN    16
#define NCONT 13
#define HDIM  400
#define K1    448      // dnn_in 429 padded to mult of 32
#define NPAD  416      // 400 padded to mult of 32 (also K of GEMM2)
#define KT1   14       // 448/32
#define KT2   13       // 416/32
#define NTILE 26       // 416/16 n-tiles

// ---------------- weight repack: f32 row-major -> bf16 WMMA-B fragment tiles ---------
// dst[((kt*NTILE+nt)*32 + L)*16 + e] = W[kt*32 + (L>>4)*8 + (e&7) + (e>>3)*16][nt*16 + (L&15)]
__global__ void dfm_pack_w(const float* __restrict__ W, int Ksrc, int Nsrc, int KT,
                           __bf16* __restrict__ dst) {
    int t = blockIdx.x * blockDim.x + threadIdx.x;
    int total = KT * NTILE * 512;
    if (t >= total) return;
    int e    = t & 15;
    int L    = (t >> 4) & 31;
    int tile = t >> 9;
    int nt   = tile % NTILE;
    int kt   = tile / NTILE;
    int k = kt * 32 + (L >> 4) * 8 + (e & 7) + (e >> 3) * 16;
    int n = nt * 16 + (L & 15);
    float v = (k < Ksrc && n < Nsrc) ? W[(size_t)k * Nsrc + n] : 0.0f;
    dst[t] = (__bf16)v;
}

// ---------------- gather + FM first/second order + bf16 dnn_in row ------------------
__global__ void dfm_gather_fm(const float* __restrict__ cont, const int* __restrict__ cat,
                              const float* __restrict__ w_cont, const float* __restrict__ b_cont,
                              const float* __restrict__ tfirst, const float* __restrict__ temb,
                              __bf16* __restrict__ A0, float* __restrict__ FM) {
    int b = blockIdx.x * blockDim.x + threadIdx.x;
    if (b >= BSZ) return;
    float se[EN], ss[EN];
#pragma unroll
    for (int e = 0; e < EN; ++e) { se[e] = 0.f; ss[e] = 0.f; }
    float fm1 = b_cont[0];
    __bf16* arow = A0 + (size_t)b * K1;
#pragma unroll
    for (int i = 0; i < NCONT; ++i) {
        float c = cont[(size_t)b * NCONT + i];
        fm1 += c * w_cont[i];
        arow[i] = (__bf16)c;
    }
    for (int f = 0; f < FN; ++f) {
        int idx = cat[(size_t)b * FN + f];
        fm1 += tfirst[(size_t)f * VN + idx];
        const float4* r = reinterpret_cast<const float4*>(temb + ((size_t)f * VN + idx) * EN);
#pragma unroll
        for (int q = 0; q < 4; ++q) {
            float4 v = r[q];
            float vv[4] = {v.x, v.y, v.z, v.w};
#pragma unroll
            for (int j = 0; j < 4; ++j) {
                int e = q * 4 + j;
                float x = vv[j];
                se[e] += x;
                ss[e] += x * x;
                arow[NCONT + f * EN + e] = (__bf16)x;
            }
        }
    }
    float fm2 = 0.f;
#pragma unroll
    for (int e = 0; e < EN; ++e) fm2 += se[e] * se[e] - ss[e];
    FM[b] = fm1 + 0.5f * fm2;
#pragma unroll
    for (int i = NCONT + FN * EN; i < K1; ++i) arow[i] = (__bf16)0.f;
}

// ---------------- WMMA bf16 GEMM + bias + ReLU, 2x2 tiles (32x32) per wave ----------
__global__ void dfm_gemm_relu(const __bf16* __restrict__ A, int lda, int KT,
                              const __bf16* __restrict__ Bp,
                              const float* __restrict__ bias, int nbias,
                              __bf16* __restrict__ C, int ldc) {
    int wave = (blockIdx.x * blockDim.x + threadIdx.x) >> 5;
    int lane = threadIdx.x & 31;
    const int NT2 = NTILE / 2;           // 13 n-pair tiles
    int nt2 = wave % NT2;
    int mt2 = wave / NT2;
    if (mt2 >= BSZ / 32) return;
    int m0 = mt2 * 32;
    int laneLo = lane & 15;
    int hi = lane >> 4;

    v8f acc00 = {}, acc01 = {}, acc10 = {}, acc11 = {};
    for (int kt = 0; kt < KT; ++kt) {
        // A fragments: lanes 0-15 cover K {0..7,16..23}, lanes 16-31 K {8..15,24..31}
        int kbase = kt * 32 + hi * 8;
        const __bf16* a0p = A + (size_t)(m0 + laneLo) * lda + kbase;
        const __bf16* a1p = A + (size_t)(m0 + 16 + laneLo) * lda + kbase;
        v16bf a0, a1, b0, b1;
        reinterpret_cast<uint4*>(&a0)[0] = *reinterpret_cast<const uint4*>(a0p);
        reinterpret_cast<uint4*>(&a0)[1] = *reinterpret_cast<const uint4*>(a0p + 16);
        reinterpret_cast<uint4*>(&a1)[0] = *reinterpret_cast<const uint4*>(a1p);
        reinterpret_cast<uint4*>(&a1)[1] = *reinterpret_cast<const uint4*>(a1p + 16);
        // B fragments: pre-packed, 16 contiguous bf16 per lane
        const __bf16* bp0 = Bp + ((size_t)(kt * NTILE + nt2 * 2) * 32 + lane) * 16;
        const __bf16* bp1 = bp0 + 512;
        reinterpret_cast<uint4*>(&b0)[0] = reinterpret_cast<const uint4*>(bp0)[0];
        reinterpret_cast<uint4*>(&b0)[1] = reinterpret_cast<const uint4*>(bp0)[1];
        reinterpret_cast<uint4*>(&b1)[0] = reinterpret_cast<const uint4*>(bp1)[0];
        reinterpret_cast<uint4*>(&b1)[1] = reinterpret_cast<const uint4*>(bp1)[1];

        acc00 = __builtin_amdgcn_wmma_f32_16x16x32_bf16(false, a0, false, b0, (short)0, acc00, false, false);
        acc01 = __builtin_amdgcn_wmma_f32_16x16x32_bf16(false, a0, false, b1, (short)0, acc01, false, false);
        acc10 = __builtin_amdgcn_wmma_f32_16x16x32_bf16(false, a1, false, b0, (short)0, acc10, false, false);
        acc11 = __builtin_amdgcn_wmma_f32_16x16x32_bf16(false, a1, false, b1, (short)0, acc11, false, false);
    }
    // epilogue: C/D layout -> VGPR i holds M = i + 8*hi, N = lane&15
    int n0 = nt2 * 32;
    int nc0 = n0 + laneLo;
    int nc1 = n0 + 16 + laneLo;
    float bv0 = (nc0 < nbias) ? bias[nc0] : 0.f;
    float bv1 = (nc1 < nbias) ? bias[nc1] : 0.f;
#pragma unroll
    for (int i = 0; i < 8; ++i) {
        int mr = i + hi * 8;
        float v;
        v = acc00[i] + bv0; v = v > 0.f ? v : 0.f; C[(size_t)(m0 + mr) * ldc + nc0]      = (__bf16)v;
        v = acc01[i] + bv1; v = v > 0.f ? v : 0.f; C[(size_t)(m0 + mr) * ldc + nc1]      = (__bf16)v;
        v = acc10[i] + bv0; v = v > 0.f ? v : 0.f; C[(size_t)(m0 + 16 + mr) * ldc + nc0] = (__bf16)v;
        v = acc11[i] + bv1; v = v > 0.f ? v : 0.f; C[(size_t)(m0 + 16 + mr) * ldc + nc1] = (__bf16)v;
    }
}

// ---------------- final fusion: out = fm*w0 + h2.Wout[1:] + b_out -------------------
__global__ void dfm_final(const __bf16* __restrict__ H2v, const float* __restrict__ FM,
                          const float* __restrict__ W_out, const float* __restrict__ b_out,
                          float* __restrict__ out) {
    int wid  = (blockIdx.x * blockDim.x + threadIdx.x) >> 5;
    int lane = threadIdx.x & 31;
    if (wid >= BSZ) return;
    float acc = 0.f;
    for (int n = lane; n < HDIM; n += 32)
        acc += (float)H2v[(size_t)wid * NPAD + n] * W_out[1 + n];
#pragma unroll
    for (int off = 16; off > 0; off >>= 1)
        acc += __shfl_xor(acc, off, 32);
    if (lane == 0)
        out[wid] = FM[wid] * W_out[0] + acc + b_out[0];
}

// ---------------- launch ------------------------------------------------------------
extern "C" void kernel_launch(void* const* d_in, const int* in_sizes, int n_in,
                              void* d_out, int out_size, void* d_ws, size_t ws_size,
                              hipStream_t stream) {
    const float* continuous = (const float*)d_in[0];
    const int*   categorical = (const int*)d_in[1];
    const float* w_cont = (const float*)d_in[2];
    const float* b_cont = (const float*)d_in[3];
    const float* tfirst = (const float*)d_in[4];
    const float* temb   = (const float*)d_in[5];
    const float* W1     = (const float*)d_in[6];
    const float* b1     = (const float*)d_in[7];
    const float* W2     = (const float*)d_in[8];
    const float* b2     = (const float*)d_in[9];
    const float* W_out  = (const float*)d_in[10];
    const float* b_out  = (const float*)d_in[11];
    float* out = (float*)d_out;

    char* ws = (char*)d_ws;
    const size_t OFF_A0 = 0;                                   // BSZ*K1 bf16   (14.68 MB)
    const size_t OFF_H1 = OFF_A0 + (size_t)BSZ * K1 * 2;       // BSZ*NPAD bf16 (13.63 MB)
    const size_t OFF_FM = OFF_H1 + (size_t)BSZ * NPAD * 2;     // BSZ f32
    const size_t OFF_W1 = OFF_FM + (size_t)BSZ * 4;            // KT1*NTILE*512 bf16
    const size_t OFF_W2 = OFF_W1 + (size_t)KT1 * NTILE * 512 * 2;

    __bf16* A0   = (__bf16*)(ws + OFF_A0);
    __bf16* H1b  = (__bf16*)(ws + OFF_H1);
    float*  FM   = (float*)(ws + OFF_FM);
    __bf16* Wb1  = (__bf16*)(ws + OFF_W1);
    __bf16* Wb2  = (__bf16*)(ws + OFF_W2);
    __bf16* H2b  = A0;   // reuse dnn_in region (dead after GEMM1)

    int pw1 = KT1 * NTILE * 512, pw2 = KT2 * NTILE * 512;
    dfm_pack_w<<<(pw1 + 255) / 256, 256, 0, stream>>>(W1, NCONT + FN * EN, HDIM, KT1, Wb1);
    dfm_pack_w<<<(pw2 + 255) / 256, 256, 0, stream>>>(W2, HDIM, HDIM, KT2, Wb2);

    dfm_gather_fm<<<BSZ / 256, 256, 0, stream>>>(continuous, categorical, w_cont, b_cont,
                                                 tfirst, temb, A0, FM);

    int waves = (BSZ / 32) * (NTILE / 2);       // 512 * 13 = 6656 waves
    int blocks = (waves * 32 + 255) / 256;      // 832 blocks of 8 waves
    dfm_gemm_relu<<<blocks, 256, 0, stream>>>(A0, K1, KT1, Wb1, b1, HDIM, H1b, NPAD);
    dfm_gemm_relu<<<blocks, 256, 0, stream>>>(H1b, NPAD, KT2, Wb2, b2, HDIM, H2b, NPAD);

    dfm_final<<<BSZ / 8, 256, 0, stream>>>(H2b, FM, W_out, b_out, out);
}
